// S2S2SModel_42666205118747
// MI455X (gfx1250) — matrix operand order repaired and checked by reading, not compiled, not run on hardware
//
#include <hip/hip_runtime.h>
#include <hip/hip_bf16.h>

typedef __attribute__((ext_vector_type(16))) __bf16 v16bf;
typedef __attribute__((ext_vector_type(8)))  float  v8f;

#define H512   512
#define G4H    2048      // 4*H
#define BATCH  256
#define TDEC   144       // 96 + 48
#define GPITCH 260       // gate staging row pitch (4*64 + 4 pad)

__device__ __forceinline__ unsigned short f2bfu(float f) {
    unsigned u = __builtin_bit_cast(unsigned, f);
    u += 0x7FFFu + ((u >> 16) & 1u);            // round-to-nearest-even
    return (unsigned short)(u >> 16);
}

__device__ __forceinline__ __bf16 f2bf(float f) {
    unsigned short s = f2bfu(f);
    return __builtin_bit_cast(__bf16, s);
}

__device__ __forceinline__ unsigned pack2bf(float lo, float hi) {
    return (unsigned)f2bfu(lo) | ((unsigned)f2bfu(hi) << 16);
}

__device__ __forceinline__ float sigf(float x) {
    return 1.0f / (1.0f + __expf(-x));
}

// ---------------------------------------------------------------- prep kernels
__global__ void k_f32_to_bf16(const float* __restrict__ src, __bf16* __restrict__ dst, int n) {
    int i = blockIdx.x * blockDim.x + threadIdx.x;
    if (i < n) dst[i] = f2bf(src[i]);
}

__global__ void k_bias_combine(const float* __restrict__ a, const float* __restrict__ b,
                               float* __restrict__ dst, int n) {
    int i = blockIdx.x * blockDim.x + threadIdx.x;
    if (i < n) dst[i] = a[i] + b[i];
}

__global__ void k_zero(float* __restrict__ dst, int n) {
    int i = blockIdx.x * blockDim.x + threadIdx.x;
    if (i < n) dst[i] = 0.0f;
}

// ---------------------------------------------------------------- fused LSTM step
// One workgroup: 32 batch rows x 64 hidden cols, all 4 gates (i,f,g,o).
// GEMM: gates[32 x 256] = [h | x_t][32 x (512+Kin)] @ W^T, bf16 WMMA, fp32 accum.
// 8 waves = 4 gates x 2 row-halves; each wave: 16 rows x 64 cols = 4 C-tiles,
// so one A-fragment feeds 4 v_wmma per K-chunk. Then the cell update for the
// workgroup's exclusive 32x64 (batch, hidden) tile.
__global__ __launch_bounds__(256) void k_lstm_step(
    const float* __restrict__ h_in, float* __restrict__ h_out,
    float* __restrict__ c_st,
    const float* __restrict__ xbase, int x_row_stride, int Kin,
    const __bf16* __restrict__ Whh,        // [2048 x 512] bf16 row-major (row = gate col)
    const __bf16* __restrict__ Wih,        // [2048 x Kin] bf16 row-major
    const float* __restrict__ bias,        // [2048] = bih + bhh
    float* __restrict__ ybase)             // yseq + t*512, row stride 144*512; null => encoder
{
    __shared__ float smem[32 * GPITCH];    // 33.3 KB: reused (A-stage bf16 | gate staging f32)
    unsigned* AsU = (unsigned*)smem;       // A-tile as dwords, pitch 17 dwords (= 34 bf16)

    const int tid  = threadIdx.x;
    const int lane = tid & 31;
    const int wave = tid >> 5;
    const int g    = wave & 3;             // gate: 0=i 1=f 2=g 3=o
    const int mh   = wave >> 2;            // 16-row half of the 32-row tile
    const int m0   = blockIdx.x * 32;      // batch tile
    const int j0   = blockIdx.y * 64;      // hidden-column tile
    const int ln   = lane & 15;
    const int half = lane >> 4;

    const int nK = (H512 + Kin) >> 5;      // K-chunks of 32 (exact: 576/544)

    v8f acc0 = {}, acc1 = {}, acc2 = {}, acc3 = {};

    const int nbase = (g << 9) + j0 + ln;  // gate column of C-tile 0 for this lane

    for (int kc = 0; kc < nK; ++kc) {
        const int kk0 = kc << 5;
        // stage A tile (32 rows x 32 k) into LDS as packed bf16 pairs:
        // 512 dwords, 2 per thread; float2 global loads, ds_store_b32 stores.
        #pragma unroll
        for (int q = 0; q < 2; ++q) {
            int p  = tid + (q << 8);          // dword index 0..511
            int r  = p >> 4;                  // row 0..31
            int kp = p & 15;                  // dword col 0..15
            int kk = kk0 + (kp << 1);
            const float* src = (kk < H512)
                ? (h_in + (size_t)(m0 + r) * H512 + kk)
                : (xbase + (size_t)(m0 + r) * x_row_stride + (kk - H512));
            float2 v2 = *(const float2*)src;
            AsU[r * 17 + kp] = pack2bf(v2.x, v2.y);
        }
        __syncthreads();

        // A fragment (16x32 bf16), ISA 7.12.2 wave32 layout
        v16bf a;
        {
            const __bf16* arow = (const __bf16*)smem + ((mh << 4) + ln) * 34;
            #pragma unroll
            for (int j = 0; j < 8; ++j) {
                int k0 = (j < 4) ? ((half << 3) + (j << 1))
                                 : (16 + (half << 3) + ((j - 4) << 1));
                a[2 * j]     = arow[k0];
                a[2 * j + 1] = arow[k0 + 1];
            }
        }

        // B fragments (32x16 bf16): B[k][n] = W[n][k], from L2-resident bf16 weights
        const __bf16* wsel;
        int ldw, koff;
        if (kk0 < H512) { wsel = Whh; ldw = H512; koff = kk0; }
        else            { wsel = Wih; ldw = Kin;  koff = kk0 - H512; }

        v16bf bfr[4];
        #pragma unroll
        for (int ct = 0; ct < 4; ++ct) {
            const __bf16* wrow = wsel + (size_t)(nbase + (ct << 4)) * ldw + koff;
            #pragma unroll
            for (int v = 0; v < 8; ++v) {
                int k = (v < 4) ? ((half << 3) + (v << 1)) : (16 + (half << 3) + ((v - 4) << 1));
                bfr[ct][2 * v]     = wrow[k];
                bfr[ct][2 * v + 1] = wrow[k + 1];
            }
        }

        acc0 = __builtin_amdgcn_wmma_f32_16x16x32_bf16(false, a, false, bfr[0],
                                                       (short)0, acc0, false, false);
        acc1 = __builtin_amdgcn_wmma_f32_16x16x32_bf16(false, a, false, bfr[1],
                                                       (short)0, acc1, false, false);
        acc2 = __builtin_amdgcn_wmma_f32_16x16x32_bf16(false, a, false, bfr[2],
                                                       (short)0, acc2, false, false);
        acc3 = __builtin_amdgcn_wmma_f32_16x16x32_bf16(false, a, false, bfr[3],
                                                       (short)0, acc3, false, false);
        __syncthreads();
    }

    // dump gates (+bias) into LDS: [32 rows][4 gates * 64 cols] (pitch GPITCH)
    {
        float b0 = bias[nbase];
        float b1 = bias[nbase + 16];
        float b2 = bias[nbase + 32];
        float b3 = bias[nbase + 48];
        #pragma unroll
        for (int r = 0; r < 8; ++r) {
            int mrow = (mh << 4) + (half << 3) + r;   // C layout: lanes16-31 hold M=r+8
            float* grow = smem + mrow * GPITCH + (g << 6) + ln;
            grow[0]  = acc0[r] + b0;
            grow[16] = acc1[r] + b1;
            grow[32] = acc2[r] + b2;
            grow[48] = acc3[r] + b3;
        }
    }
    __syncthreads();

    // LSTM cell update for the exclusive 32x64 (batch, hidden) tile
    #pragma unroll
    for (int q = 0; q < 8; ++q) {
        int idx = tid + (q << 8);
        int r = idx >> 6, jj = idx & 63;
        const float* grow = smem + r * GPITCH;
        float gi = grow[jj];
        float gf = grow[64 + jj];
        float gg = grow[128 + jj];
        float go = grow[192 + jj];
        size_t hoff = (size_t)(m0 + r) * H512 + (j0 + jj);
        float cp = c_st[hoff];
        float cn = sigf(gf) * cp + sigf(gi) * tanhf(gg);
        float hn = sigf(go) * tanhf(cn);
        c_st[hoff]  = cn;
        h_out[hoff] = hn;
        if (ybase) ybase[(size_t)(m0 + r) * (TDEC * H512) + (j0 + jj)] = hn;
    }
}

// ---------------------------------------------------------------- head: out[b,t] = y . W + b
__global__ __launch_bounds__(256) void k_head(const float* __restrict__ yseq,
                                              const float* __restrict__ W,
                                              const float* __restrict__ bvec,
                                              float* __restrict__ out, int BT) {
    int widx = (blockIdx.x * blockDim.x + threadIdx.x) >> 5;  // one wave32 per output
    int lane = threadIdx.x & 31;
    if (widx >= BT) return;
    const float* row = yseq + (size_t)widx * H512;
    float s = 0.0f;
    #pragma unroll 4
    for (int j = lane; j < H512; j += 32) s += row[j] * W[j];
    #pragma unroll
    for (int off = 16; off; off >>= 1) s += __shfl_down(s, off, 32);
    if (lane == 0) out[widx] = s + bvec[0];
}

// ---------------------------------------------------------------- launch
extern "C" void kernel_launch(void* const* d_in, const int* in_sizes, int n_in,
                              void* d_out, int out_size, void* d_ws, size_t ws_size,
                              hipStream_t stream) {
    const float* x_d       = (const float*)d_in[0];
    const float* x_f       = (const float*)d_in[1];
    const float* x_ff      = (const float*)d_in[2];
    const float* enc_Wih   = (const float*)d_in[3];
    const float* enc_Whh   = (const float*)d_in[4];
    const float* enc_bih   = (const float*)d_in[5];
    const float* enc_bhh   = (const float*)d_in[6];
    const float* decf_Wih  = (const float*)d_in[7];
    const float* decf_Whh  = (const float*)d_in[8];
    const float* decf_bih  = (const float*)d_in[9];
    const float* decf_bhh  = (const float*)d_in[10];
    const float* decff_Wih = (const float*)d_in[11];
    const float* decff_Whh = (const float*)d_in[12];
    const float* decff_bih = (const float*)d_in[13];
    const float* decff_bhh = (const float*)d_in[14];
    const float* head_W    = (const float*)d_in[15];
    const float* head_b    = (const float*)d_in[16];
    float* out = (float*)d_out;

    // workspace layout (floats / bf16), all 4-byte aligned
    float* ws        = (float*)d_ws;
    float* h0        = ws;                           // 256*512
    float* h1        = h0 + BATCH * H512;
    float* cst       = h1 + BATCH * H512;
    float* encBias   = cst + BATCH * H512;           // 2048
    float* dfBias    = encBias + G4H;
    float* dffBias   = dfBias + G4H;
    __bf16* encWhh   = (__bf16*)(dffBias + G4H);     // 2048*512
    __bf16* encWih   = encWhh + (size_t)G4H * H512;  // 2048*64
    __bf16* dfWhh    = encWih + (size_t)G4H * 64;
    __bf16* dfWih    = dfWhh + (size_t)G4H * H512;   // 2048*32
    __bf16* dffWhh   = dfWih + (size_t)G4H * 32;
    __bf16* dffWih   = dffWhh + (size_t)G4H * H512;
    float* yseq      = (float*)(dffWih + (size_t)G4H * 32);   // 256*144*512

    auto cvt = [&](const float* s, __bf16* d, int n) {
        k_f32_to_bf16<<<(n + 255) / 256, 256, 0, stream>>>(s, d, n);
    };
    cvt(enc_Whh,   encWhh, G4H * H512);
    cvt(enc_Wih,   encWih, G4H * 64);
    cvt(decf_Whh,  dfWhh,  G4H * H512);
    cvt(decf_Wih,  dfWih,  G4H * 32);
    cvt(decff_Whh, dffWhh, G4H * H512);
    cvt(decff_Wih, dffWih, G4H * 32);
    k_bias_combine<<<(G4H + 255) / 256, 256, 0, stream>>>(enc_bih,   enc_bhh,   encBias, G4H);
    k_bias_combine<<<(G4H + 255) / 256, 256, 0, stream>>>(decf_bih,  decf_bhh,  dfBias,  G4H);
    k_bias_combine<<<(G4H + 255) / 256, 256, 0, stream>>>(decff_bih, decff_bhh, dffBias, G4H);
    k_zero<<<(BATCH * H512 + 255) / 256, 256, 0, stream>>>(h0,  BATCH * H512);
    k_zero<<<(BATCH * H512 + 255) / 256, 256, 0, stream>>>(cst, BATCH * H512);

    dim3 sgrid(BATCH / 32, H512 / 64);   // 8 x 8 workgroups
    float* hc = h0;
    float* hn = h1;

    // encoder: 336 steps, Kin=64
    for (int t = 0; t < 336; ++t) {
        k_lstm_step<<<sgrid, 256, 0, stream>>>(hc, hn, cst, x_d + (size_t)t * 64, 336 * 64, 64,
                                               encWhh, encWih, encBias, (float*)nullptr);
        float* tmp = hc; hc = hn; hn = tmp;
    }
    // forecast decoder: 96 steps, Kin=32, writes yseq[:, 0:96, :]
    for (int t = 0; t < 96; ++t) {
        k_lstm_step<<<sgrid, 256, 0, stream>>>(hc, hn, cst, x_f + (size_t)t * 32, 96 * 32, 32,
                                               dfWhh, dfWih, dfBias, yseq + (size_t)t * H512);
        float* tmp = hc; hc = hn; hn = tmp;
    }
    // future decoder: 48 steps, Kin=32, writes yseq[:, 96:144, :]
    for (int t = 0; t < 48; ++t) {
        k_lstm_step<<<sgrid, 256, 0, stream>>>(hc, hn, cst, x_ff + (size_t)t * 32, 48 * 32, 32,
                                               dffWhh, dffWih, dffBias, yseq + (size_t)(96 + t) * H512);
        float* tmp = hc; hc = hn; hn = tmp;
    }

    // head: [256*144] outputs, one wave per output
    int BT = BATCH * TDEC;
    k_head<<<(BT * 32 + 255) / 256, 256, 0, stream>>>(yseq, head_W, head_b, out, BT);
}
